// MixtureOfExperts_55843164782858
// MI455X (gfx1250) — matrix operand order repaired and checked by reading, not compiled, not run on hardware
//
#include <hip/hip_runtime.h>
#include <hip/hip_bf16.h>
#include <stdint.h>

#define B_TOK   8192
#define D_IN    1024
#define D_HID   2048
#define D_OUT   1024
#define NEXP    8

typedef __attribute__((ext_vector_type(16))) __bf16          v16bf;
typedef __attribute__((ext_vector_type(4)))  __bf16          v4bf;
typedef __attribute__((ext_vector_type(2)))  __bf16          v2bf;
typedef __attribute__((ext_vector_type(8)))  float           v8f;
typedef __attribute__((ext_vector_type(8)))  unsigned short  v8u;
typedef __attribute__((ext_vector_type(4)))  unsigned int    v4u32;
typedef __attribute__((ext_vector_type(8)))  int             v8i32;
typedef __attribute__((ext_vector_type(4)))  int             v4i32;

#if __has_builtin(__builtin_amdgcn_tensor_load_to_lds) && \
    __has_builtin(__builtin_amdgcn_s_wait_tensorcnt)
#define USE_TDM 1
#else
#define USE_TDM 0
#endif

// hardware bf16 convert (v_cvt_pk_bf16_f32 on gfx1250)
__device__ __forceinline__ unsigned short f2bf(float f) {
  union { __bf16 h; unsigned short u; } c;
  c.h = (__bf16)f;
  return c.u;
}
// pack two f32 -> <2 x bf16> dword store (fuses into one v_cvt_pk_bf16_f32)
__device__ __forceinline__ void store_pk(unsigned short* p, float lo, float hi) {
  v2bf t;
  t.x = (__bf16)lo;
  t.y = (__bf16)hi;
  *reinterpret_cast<v2bf*>(p) = t;
}

union FragBF { v8u u[2]; v16bf b; };

// ---------------- fp32 -> bf16 bulk convert (x) ----------------
__global__ __launch_bounds__(256) void cvt_f32_bf16(const float* __restrict__ src,
                                                    unsigned short* __restrict__ dst,
                                                    int n) {
  int i = (blockIdx.x * 256 + threadIdx.x) * 4;
  if (i + 3 < n) {
    const float4 v = *reinterpret_cast<const float4*>(src + i);
    v4bf o;
    o.x = (__bf16)v.x; o.y = (__bf16)v.y; o.z = (__bf16)v.z; o.w = (__bf16)v.w;
    *reinterpret_cast<v4bf*>(dst + i) = o;
  }
}

// ---------------- gating: softmax(x @ gate_w + gate_b) ----------------
__global__ __launch_bounds__(256) void moe_gates(const float* __restrict__ x,
                                                 const float* __restrict__ gw,
                                                 const float* __restrict__ gb,
                                                 float* __restrict__ gates) {
  const int wave = threadIdx.x >> 5;
  const int lane = threadIdx.x & 31;
  const int row  = blockIdx.x * 8 + wave;

  float acc[NEXP];
#pragma unroll
  for (int e = 0; e < NEXP; ++e) acc[e] = 0.f;

  const float* xr = x + (size_t)row * D_IN;
  for (int k = lane; k < D_IN; k += 32) {
    const float xv = xr[k];
    const float* g = gw + (size_t)k * NEXP;
#pragma unroll
    for (int e = 0; e < NEXP; ++e) acc[e] += xv * g[e];
  }
#pragma unroll
  for (int off = 16; off > 0; off >>= 1) {
#pragma unroll
    for (int e = 0; e < NEXP; ++e) acc[e] += __shfl_xor(acc[e], off, 32);
  }
  float mx = -3.4e38f;
#pragma unroll
  for (int e = 0; e < NEXP; ++e) { acc[e] += gb[e]; mx = fmaxf(mx, acc[e]); }
  float s = 0.f;
#pragma unroll
  for (int e = 0; e < NEXP; ++e) { acc[e] = __expf(acc[e] - mx); s += acc[e]; }
  const float inv = 1.f / s;
  if (lane == 0) {
#pragma unroll
    for (int e = 0; e < NEXP; ++e) gates[(size_t)row * NEXP + e] = acc[e] * inv;
  }
}

#if USE_TDM
// TDM 2D tile load: 256 rows x 32 cols of bf16 (2B) -> LDS, row-major.
// D# packing per CDNA5 ISA (groups 0/1; groups 2/3 zero: 2D).
__device__ __forceinline__ void tdm_load_tile(const unsigned short* gptr,
                                              unsigned int lds_byte_off,
                                              int row_stride_elems) {
  const unsigned long long ga = (unsigned long long)(uintptr_t)gptr;
  v4u32 g0;
  g0.x = 1u;                                             // count=1, user mode
  g0.y = lds_byte_off;                                   // lds_addr
  g0.z = (unsigned int)(ga & 0xFFFFFFFFull);             // global_addr[31:0]
  g0.w = (unsigned int)((ga >> 32) & 0x01FFFFFFull)      // global_addr[56:32]
         | (2u << 30);                                   // type=2 (image)
  v8i32 g1;
  const unsigned int td0 = (unsigned int)row_stride_elems;   // tensor_dim0
  const unsigned int td1 = (unsigned int)B_TOK;              // tensor_dim1
  const unsigned long long s0 = (unsigned long long)row_stride_elems;
  g1[0] = (int)(1u << 16);                               // data_size = 2 bytes
  g1[1] = (int)((td0 & 0xFFFFu) << 16);                  // tensor_dim0[15:0]
  g1[2] = (int)((td0 >> 16) | ((td1 & 0xFFFFu) << 16));  // td0[31:16] | td1[15:0]
  g1[3] = (int)((td1 >> 16) | (32u << 16));              // td1[31:16] | tile_dim0=32
  g1[4] = (int)256u;                                     // tile_dim1 = 256 rows
  g1[5] = (int)(unsigned int)(s0 & 0xFFFFFFFFull);       // tensor_dim0_stride lo
  g1[6] = (int)(unsigned int)((s0 >> 32) & 0xFFFFull);   // stride hi
  g1[7] = 0;
  v4i32 z4 = {0, 0, 0, 0};
  v8i32 z8 = {0, 0, 0, 0, 0, 0, 0, 0};
  __builtin_amdgcn_tensor_load_to_lds(g0, g1, z4, z4, z8, 0);
}
#endif

// ---------------- bf16 WMMA GEMM ----------------
// Block tile 256x128, BK=32. 8 waves in 4x2 grid, each wave 64x64 via 4x4
// grid of 16x16x32 bf16 WMMA tiles (acc = 128 VGPRs).
// MODE 0: relu -> store bf16      MODE 1: gate-weighted accumulate into f32
template <int MODE>
__global__ __launch_bounds__(256) void moe_gemm_bf16(
    const unsigned short* __restrict__ A, const float* __restrict__ W,
    const float* __restrict__ bias, unsigned short* __restrict__ Cb,
    float* __restrict__ Cf, const float* __restrict__ gates,
    int M, int N, int K, int expert, int accumulate) {
  // ping-pong: A0 A1 (256x32 = 16KB each), B0 B1 (128x32 [n][k] = 8KB each)
  __shared__ __align__(16) unsigned short smem[2 * 8192 + 2 * 4096];

  const int tid  = threadIdx.x;
  const int lane = tid & 31;
  const int wave = tid >> 5;
  const int wr   = wave >> 1;   // 0..3 -> 64-row slab
  const int wc   = wave & 1;    // 0..1 -> 64-col slab
  const int lh   = lane >> 4;
  const int ll   = lane & 15;
  const int m0   = blockIdx.y * 256;
  const int n0   = blockIdx.x * 128;

  auto Abuf = [&](int i) -> unsigned short* { return smem + i * 8192; };
  auto Bbuf = [&](int i) -> unsigned short* { return smem + 16384 + i * 4096; };

  v8f acc[4][4];
#pragma unroll
  for (int i = 0; i < 4; ++i)
#pragma unroll
    for (int j = 0; j < 4; ++j)
#pragma unroll
      for (int v = 0; v < 8; ++v) acc[i][j][v] = 0.f;

  const int ksteps = K >> 5;

  // --- B tile staging: 32x128 f32; thread takes a (k,k+1) row pair x 4 cols ---
  float4 breg[4];
  auto loadB = [&](int k0) {
#pragma unroll
    for (int j = 0; j < 2; ++j) {
      const int v  = tid + j * 256;
      const int kk = (v >> 5) << 1;
      const int nn = (v & 31) << 2;
      breg[2 * j + 0] = *reinterpret_cast<const float4*>(W + (size_t)(k0 + kk) * N + n0 + nn);
      breg[2 * j + 1] = *reinterpret_cast<const float4*>(W + (size_t)(k0 + kk + 1) * N + n0 + nn);
    }
  };
  auto storeB = [&](int buf) {
    unsigned short* bs = Bbuf(buf);
#pragma unroll
    for (int j = 0; j < 2; ++j) {
      const int v  = tid + j * 256;
      const int kk = (v >> 5) << 1;
      const int nn = (v & 31) << 2;
      const float4 lo = breg[2 * j + 0];
      const float4 hi = breg[2 * j + 1];
      // [n][k]: consecutive k adjacent -> one cvt_pk + b32 store per (n, k-pair)
      store_pk(&bs[(nn + 0) * 32 + kk], lo.x, hi.x);
      store_pk(&bs[(nn + 1) * 32 + kk], lo.y, hi.y);
      store_pk(&bs[(nn + 2) * 32 + kk], lo.z, hi.z);
      store_pk(&bs[(nn + 3) * 32 + kk], lo.w, hi.w);
    }
  };

#if USE_TDM
  auto issueA = [&](int k0, int buf) {
    if (wave == 0)
      tdm_load_tile(A + (size_t)m0 * K + k0, (unsigned int)(buf * 16384), K);
  };
#else
  v8u areg[4];
  auto issueA = [&](int k0, int buf) {
#pragma unroll
    for (int j = 0; j < 4; ++j) {
      const int v   = tid + j * 256;
      const int row = v >> 2;
      const int col = (v & 3) << 3;
      areg[j] = *reinterpret_cast<const v8u*>(A + (size_t)(m0 + row) * K + k0 + col);
    }
    unsigned short* as = Abuf(buf);
#pragma unroll
    for (int j = 0; j < 4; ++j) {
      const int v   = tid + j * 256;
      const int row = v >> 2;
      const int col = (v & 3) << 3;
      *reinterpret_cast<v8u*>(&as[row * 32 + col]) = areg[j];
    }
  };
#endif

  // ---- prologue: tile 0 ----
  issueA(0, 0);
  loadB(0);
  storeB(0);
#if USE_TDM
  if (wave == 0) __builtin_amdgcn_s_wait_tensorcnt(0);
#endif
  __syncthreads();

  for (int ks = 0; ks < ksteps; ++ks) {
    const int cur  = ks & 1;
    const int nxt  = cur ^ 1;
    const bool more = (ks + 1 < ksteps);

    if (more) {
      issueA((ks + 1) << 5, nxt);   // TDM DMA overlaps WMMA compute
      loadB((ks + 1) << 5);         // global loads in flight during WMMA
      __builtin_prefetch(W + (size_t)((ks + 1) << 5) * N + n0 + tid, 0, 1);
    }

    // ---- compute from buffers[cur]: 16 WMMA per wave per K-step ----
    const unsigned short* as = Abuf(cur);
    const unsigned short* bs = Bbuf(cur);
    FragBF bfrag[4];
#pragma unroll
    for (int nt = 0; nt < 4; ++nt) {
      const unsigned short* p = &bs[(wc * 64 + nt * 16 + ll) * 32 + lh * 8];
      bfrag[nt].u[0] = *reinterpret_cast<const v8u*>(p);
      bfrag[nt].u[1] = *reinterpret_cast<const v8u*>(p + 16);
    }
#pragma unroll
    for (int mt = 0; mt < 4; ++mt) {
      FragBF afrag;
      const unsigned short* p = &as[(wr * 64 + mt * 16 + ll) * 32 + lh * 8];
      afrag.u[0] = *reinterpret_cast<const v8u*>(p);
      afrag.u[1] = *reinterpret_cast<const v8u*>(p + 16);
#pragma unroll
      for (int nt = 0; nt < 4; ++nt) {
        acc[mt][nt] = __builtin_amdgcn_wmma_f32_16x16x32_bf16(
            false, afrag.b, false, bfrag[nt].b, (short)0, acc[mt][nt], false, false);
      }
    }

    if (more) {
      storeB(nxt);
#if USE_TDM
      if (wave == 0) __builtin_amdgcn_s_wait_tensorcnt(0);
#endif
    }
    __syncthreads();
  }

  // ---- epilogue: D layout — lane ll = col, VGPR v -> row v + 8*lh ----
#pragma unroll
  for (int mt = 0; mt < 4; ++mt) {
#pragma unroll
    for (int nt = 0; nt < 4; ++nt) {
      const int n  = n0 + wc * 64 + nt * 16 + ll;
      const float bn = bias[n];
#pragma unroll
      for (int v = 0; v < 8; ++v) {
        const int m = m0 + wr * 64 + mt * 16 + lh * 8 + v;
        float r = acc[mt][nt][v] + bn;
        if (MODE == 0) {
          r = r > 0.f ? r : 0.f;
          Cb[(size_t)m * N + n] = f2bf(r);
        } else {
          const float g = gates[(size_t)m * NEXP + expert];
          float o = g * r;
          const size_t idx = (size_t)m * N + n;
          if (accumulate) o += Cf[idx];
          Cf[idx] = o;
        }
      }
    }
  }
}

static inline size_t alignup(size_t x) { return (x + 255) & ~(size_t)255; }

extern "C" void kernel_launch(void* const* d_in, const int* in_sizes, int n_in,
                              void* d_out, int out_size, void* d_ws, size_t ws_size,
                              hipStream_t stream) {
  const float* x  = (const float*)d_in[0];
  const float* gw = (const float*)d_in[1];
  const float* gb = (const float*)d_in[2];
  const float* w1 = (const float*)d_in[3];
  const float* b1 = (const float*)d_in[4];
  const float* w2 = (const float*)d_in[5];
  const float* b2 = (const float*)d_in[6];
  const float* w3 = (const float*)d_in[7];
  const float* b3 = (const float*)d_in[8];
  float* out = (float*)d_out;

  char* ws = (char*)d_ws;
  float* gates = (float*)ws;                 ws += alignup((size_t)B_TOK * NEXP * 4);
  unsigned short* xb  = (unsigned short*)ws; ws += alignup((size_t)B_TOK * D_IN * 2);
  unsigned short* h1b = (unsigned short*)ws; ws += alignup((size_t)B_TOK * D_HID * 2);
  unsigned short* h2b = (unsigned short*)ws; ws += alignup((size_t)B_TOK * D_HID * 2);

  cvt_f32_bf16<<<(B_TOK * D_IN / 4 + 255) / 256, 256, 0, stream>>>(x, xb, B_TOK * D_IN);
  moe_gates<<<B_TOK / 8, 256, 0, stream>>>(x, gw, gb, gates);

  for (int e = 0; e < NEXP; ++e) {
    moe_gemm_bf16<0><<<dim3(D_HID / 128, B_TOK / 256), 256, 0, stream>>>(
        xb, w1 + (size_t)e * D_IN * D_HID, b1 + (size_t)e * D_HID,
        h1b, nullptr, nullptr, B_TOK, D_HID, D_IN, e, 0);
    moe_gemm_bf16<0><<<dim3(D_HID / 128, B_TOK / 256), 256, 0, stream>>>(
        h1b, w2 + (size_t)e * D_HID * D_HID, b2 + (size_t)e * D_HID,
        h2b, nullptr, nullptr, B_TOK, D_HID, D_HID, e, 0);
    moe_gemm_bf16<1><<<dim3(D_OUT / 128, B_TOK / 256), 256, 0, stream>>>(
        h2b, w3 + (size_t)e * D_HID * D_OUT, b3 + (size_t)e * D_OUT,
        nullptr, out, gates, B_TOK, D_OUT, D_HID, e, e != 0);
  }
}
